// NaryTreeLSTMCell_72550587564075
// MI455X (gfx1250) — matrix operand order repaired and checked by reading, compiled, not run on hardware
//
#include <hip/hip_runtime.h>
#include <stdint.h>

// ---------------------------------------------------------------------------
// NaryTreeLSTMCell for MI455X (gfx1250, wave32, WMMA bf16 -> f32 accumulate)
// ---------------------------------------------------------------------------

typedef __attribute__((ext_vector_type(16))) __bf16 v16bf;
typedef __attribute__((ext_vector_type(8)))  float  v8f;
typedef __attribute__((ext_vector_type(4)))  int    v4i;

#define BATCH 32
#define LSEQ  512
#define BL    16384        // BATCH * LSEQ rows
#define KDIM  1024         // E == H
#define NDIM  4096         // concatenated output columns
#define HDIM  1024

// GEMM tiling
#define BM 128
#define BN 128
#define BK 32
#define PAD 8              // LDS row padding (bf16 elems) to spread banks

// Prefer the CDNA5 async global->LDS path when the toolchain exposes it;
// fall back to VGPR-staged copies otherwise (both double-buffered).
#ifndef USE_ASYNC
#define USE_ASYNC 1
#endif
#if USE_ASYNC && defined(__has_builtin)
#if __has_builtin(__builtin_amdgcn_global_load_async_to_lds_b128)
#define ASYNC_COPY 1
#endif
#endif
#ifndef ASYNC_COPY
#define ASYNC_COPY 0
#endif

static __device__ __forceinline__ unsigned short f2bf(float f) {
  union { float f; uint32_t u; } v; v.f = f;
  uint32_t r = v.u + 0x7FFFu + ((v.u >> 16) & 1u);   // round-to-nearest-even
  return (unsigned short)(r >> 16);
}
static __device__ __forceinline__ float sigmoidf_(float x) {
  return 1.0f / (1.0f + __expf(-x));
}

#if ASYNC_COPY
static __device__ __forceinline__ void async_cp16(const unsigned short* g,
                                                  unsigned short* l) {
  // prototype (from hipcc diagnostic): (v4i AS1*, v4i AS3*, Ii offset, Ii cpol)
  __builtin_amdgcn_global_load_async_to_lds_b128(
      (__attribute__((address_space(1))) v4i*)(unsigned short*)g,
      (__attribute__((address_space(3))) v4i*)l, 0, 0);
}
static __device__ __forceinline__ void wait_async0() {
#if __has_builtin(__builtin_amdgcn_s_wait_asynccnt)
  __builtin_amdgcn_s_wait_asynccnt(0);
#else
  asm volatile("s_wait_asynccnt 0x0" ::: "memory");
#endif
}
#endif

// ---------------------------------------------------------------------------
// Weight packing: column-major bf16 [N][K] so B-tile loads are contiguous.
//  Wpx: cols 0..3071 = W_ioux, cols 3072..4095 = W_fx
//  Wph: cols 0..1023 = W_iouh[0][:, :H], 1024..2047 = W_iouh[1][:, :H],
//       2048..3071 = W_fh[0]+W_fh[1],    3072..4095 = W_fh[2]+W_fh[3]
// ---------------------------------------------------------------------------
__global__ __launch_bounds__(256) void pack_weights_kernel(
    const float* __restrict__ W_ioux, const float* __restrict__ W_iouh,
    const float* __restrict__ W_fx,   const float* __restrict__ W_fh,
    unsigned short* __restrict__ Wpx, unsigned short* __restrict__ Wph)
{
  int idx = blockIdx.x * 256 + threadIdx.x;   // over NDIM*KDIM
  int n = idx >> 10;
  int k = idx & 1023;

  float wx = (n < 3072) ? W_ioux[(size_t)k * 3072 + n]
                        : W_fx[(size_t)k * 1024 + (n - 3072)];
  float wh;
  if (n < 1024) {
    wh = W_iouh[(size_t)k * 3072 + n];
  } else if (n < 2048) {
    wh = W_iouh[(size_t)3072 * 1024 + (size_t)k * 3072 + (n - 1024)];
  } else if (n < 3072) {
    int np = n - 2048;
    wh = W_fh[(size_t)k * 1024 + np] + W_fh[(size_t)1048576 + (size_t)k * 1024 + np];
  } else {
    int np = n - 3072;
    wh = W_fh[(size_t)2097152 + (size_t)k * 1024 + np] +
         W_fh[(size_t)3145728 + (size_t)k * 1024 + np];
  }
  Wpx[(size_t)n * KDIM + k] = f2bf(wx);
  Wph[(size_t)n * KDIM + k] = f2bf(wh);
}

__global__ __launch_bounds__(256) void pack_bias_kernel(
    const float* __restrict__ b_iouh, const float* __restrict__ b_fh,
    float* __restrict__ biasN)
{
  int n = blockIdx.x * 256 + threadIdx.x;
  float v;
  if (n < 1024)       v = b_iouh[n];
  else if (n < 2048)  v = b_iouh[3072 + (n - 1024)];
  else if (n < 3072) { int np = n - 2048; v = b_fh[np] + b_fh[1024 + np]; }
  else               { int np = n - 3072; v = b_fh[2048 + np] + b_fh[3072 + np]; }
  biasN[n] = v;
}

// Convert x and h0 (fp32) to bf16 activation buffers, 4 elems/thread.
__global__ __launch_bounds__(256) void convert_bf16_kernel(
    const float* __restrict__ x, const float* __restrict__ h0,
    unsigned short* __restrict__ xb, unsigned short* __restrict__ hb)
{
  size_t i = ((size_t)blockIdx.x * 256 + threadIdx.x) * 4;
  const float4 xv = *(const float4*)(x + i);
  const float4 hv = *(const float4*)(h0 + i);
  ushort4 xo, ho;
  xo.x = f2bf(xv.x); xo.y = f2bf(xv.y); xo.z = f2bf(xv.z); xo.w = f2bf(xv.w);
  ho.x = f2bf(hv.x); ho.y = f2bf(hv.y); ho.z = f2bf(hv.z); ho.w = f2bf(hv.w);
  *(ushort4*)(xb + i) = xo;
  *(ushort4*)(hb + i) = ho;
}

__global__ __launch_bounds__(256) void zero_kernel(float* __restrict__ p, size_t n4)
{
  size_t i = (size_t)blockIdx.x * 256 + threadIdx.x;
  if (i < n4) ((float4*)p)[i] = make_float4(0.f, 0.f, 0.f, 0.f);
}

// ---------------------------------------------------------------------------
// bf16 WMMA GEMM: C[BLxNDIM] = A[BLxKDIM] * Bw^T  (Bw stored [N][K] col-major)
// Block 128x128x32, 8 waves, each wave owns 64x32 (4x2 wmma accumulators).
// Double-buffered LDS: one barrier per K-step; next tile's copies overlap the
// current tile's 8 WMMAs (async global->LDS when available).
// mode 0: plain f32 store to outx.
// mode 1: val += biasN[n]; n<1024  -> atomic scatter via idx_r into iou_acc
//                          n<2048  -> atomic scatter via idx_l into iou_acc
//                          else    -> store to frfl (fr | fl)
// ---------------------------------------------------------------------------
__global__ __launch_bounds__(256) void gemm_wmma_kernel(
    const unsigned short* __restrict__ A,
    const unsigned short* __restrict__ Bw,
    float* __restrict__ outx,
    const float* __restrict__ biasN,
    const int* __restrict__ idx_r,
    const int* __restrict__ idx_l,
    float* __restrict__ iou_acc,
    float* __restrict__ frfl,
    int mode)
{
  __shared__ unsigned short As[2][BM][BK + PAD];
  __shared__ unsigned short Bs[2][BN][BK + PAD];

  const int tid   = threadIdx.x;
  const int lane  = tid & 31;
  const int wave  = tid >> 5;
  const int wm    = (wave & 1) * 64;   // wave row base within block tile
  const int wn    = (wave >> 1) * 32;  // wave col base within block tile
  const int bm0   = blockIdx.y * BM;
  const int bn0   = blockIdx.x * BN;
  const int l16   = lane & 15;
  const int khalf = lane >> 4;

  // per-thread tile-copy coordinates (2 x 16B for A, 2 x 16B for B)
  int lrow[2], lc8[2];
#pragma unroll
  for (int p = 0; p < 2; p++) {
    int idx = tid + p * 256;
    lrow[p] = idx >> 2;
    lc8[p]  = (idx & 3) * 8;
  }

  v8f acc[4][2];
  const v8f vzero = {0.f, 0.f, 0.f, 0.f, 0.f, 0.f, 0.f, 0.f};
#pragma unroll
  for (int i = 0; i < 4; i++)
#pragma unroll
    for (int j = 0; j < 2; j++) acc[i][j] = vzero;

#if ASYNC_COPY
  // ---- prologue: async copy tile k0=0 into buffer 0
#pragma unroll
  for (int p = 0; p < 2; p++) {
    async_cp16(A  + (size_t)(bm0 + lrow[p]) * KDIM + lc8[p], &As[0][lrow[p]][lc8[p]]);
    async_cp16(Bw + (size_t)(bn0 + lrow[p]) * KDIM + lc8[p], &Bs[0][lrow[p]][lc8[p]]);
  }
  wait_async0();
  __syncthreads();
#else
  uint4 ra[2], rb[2];
#pragma unroll
  for (int p = 0; p < 2; p++) {
    ra[p] = *(const uint4*)(A  + (size_t)(bm0 + lrow[p]) * KDIM + lc8[p]);
    rb[p] = *(const uint4*)(Bw + (size_t)(bn0 + lrow[p]) * KDIM + lc8[p]);
  }
#pragma unroll
  for (int p = 0; p < 2; p++) {
    *(uint4*)(&As[0][lrow[p]][lc8[p]]) = ra[p];
    *(uint4*)(&Bs[0][lrow[p]][lc8[p]]) = rb[p];
  }
  __syncthreads();
#endif

  for (int k0 = 0; k0 < KDIM; k0 += BK) {
    const int cur = (k0 >> 5) & 1;
    const int nxt = cur ^ 1;
    const bool pf = (k0 + BK) < KDIM;

    // ---- kick off next tile's copies before computing on the current one
#if ASYNC_COPY
    if (pf) {
      int kn = k0 + BK;
#pragma unroll
      for (int p = 0; p < 2; p++) {
        async_cp16(A  + (size_t)(bm0 + lrow[p]) * KDIM + kn + lc8[p],
                   &As[nxt][lrow[p]][lc8[p]]);
        async_cp16(Bw + (size_t)(bn0 + lrow[p]) * KDIM + kn + lc8[p],
                   &Bs[nxt][lrow[p]][lc8[p]]);
      }
    }
#else
    if (pf) {
      int kn = k0 + BK;
#pragma unroll
      for (int p = 0; p < 2; p++) {
        ra[p] = *(const uint4*)(A  + (size_t)(bm0 + lrow[p]) * KDIM + kn + lc8[p]);
        rb[p] = *(const uint4*)(Bw + (size_t)(bn0 + lrow[p]) * KDIM + kn + lc8[p]);
      }
    }
#endif

    // ---- 8 WMMAs on the current buffer
    union Frag { uint4 q[2]; v16bf v; };
    Frag a[4], b[2];
    // A 16x32 bf16 layout: lanes 0-15 row M=l16 hold K {0..7,16..23},
    // lanes 16-31 hold K {8..15,24..31}  -> two 16B LDS reads per frag.
#pragma unroll
    for (int mt = 0; mt < 4; mt++) {
      int row = wm + mt * 16 + l16;
      a[mt].q[0] = *(const uint4*)(&As[cur][row][8 * khalf]);
      a[mt].q[1] = *(const uint4*)(&As[cur][row][16 + 8 * khalf]);
    }
    // B 32x16 bf16 layout: lane holds column N=l16, K {0..7,16..23} (lanes 0-15) / {8..15,24..31} (lanes 16-31).
#pragma unroll
    for (int nt = 0; nt < 2; nt++) {
      int col = wn + nt * 16 + l16;
      b[nt].q[0] = *(const uint4*)(&Bs[cur][col][8 * khalf]);
      b[nt].q[1] = *(const uint4*)(&Bs[cur][col][16 + 8 * khalf]);
    }
#pragma unroll
    for (int mt = 0; mt < 4; mt++)
#pragma unroll
      for (int nt = 0; nt < 2; nt++)
        acc[mt][nt] = __builtin_amdgcn_wmma_f32_16x16x32_bf16(
            false, a[mt].v, false, b[nt].v, (short)0, acc[mt][nt], false, false);

    // ---- finish next tile (waits only after compute has issued)
#if ASYNC_COPY
    if (pf) wait_async0();
#else
    if (pf) {
#pragma unroll
      for (int p = 0; p < 2; p++) {
        *(uint4*)(&As[nxt][lrow[p]][lc8[p]]) = ra[p];
        *(uint4*)(&Bs[nxt][lrow[p]][lc8[p]]) = rb[p];
      }
    }
#endif
    __syncthreads();
  }

  // Epilogue. C/D layout: lane -> N=l16; vgpr e -> M = e + 8*khalf.
  if (mode == 0) {
#pragma unroll
    for (int mt = 0; mt < 4; mt++)
#pragma unroll
      for (int nt = 0; nt < 2; nt++) {
        int n = bn0 + wn + nt * 16 + l16;
#pragma unroll
        for (int e = 0; e < 8; e++) {
          int m = bm0 + wm + mt * 16 + e + 8 * khalf;
          outx[(size_t)m * NDIM + n] = acc[mt][nt][e];
        }
      }
  } else {
#pragma unroll
    for (int mt = 0; mt < 4; mt++)
#pragma unroll
      for (int nt = 0; nt < 2; nt++) {
        int n = bn0 + wn + nt * 16 + l16;
        float bias = biasN[n];
#pragma unroll
        for (int e = 0; e < 8; e++) {
          int m = bm0 + wm + mt * 16 + e + 8 * khalf;
          float val = acc[mt][nt][e] + bias;
          int rowbase = (m >> 9) << 9;          // b * L
          if (n < HDIM) {
            int tgt = idx_r[m];
            atomicAdd(&iou_acc[(size_t)(rowbase + tgt) * HDIM + n], val);
          } else if (n < 2 * HDIM) {
            int tgt = idx_l[m];
            atomicAdd(&iou_acc[(size_t)(rowbase + tgt) * HDIM + (n - HDIM)], val);
          } else {
            frfl[(size_t)m * 2048 + (n - 2048)] = val;
          }
        }
      }
  }
}

// ---------------------------------------------------------------------------
// Row compaction map for masked_scatter semantics: the k-th True row (flat
// order, mask = tree_ids_d != 0) receives h_new/c_new row k.
// Single block, 512 threads x 32 rows, Hillis-Steele scan.
// ---------------------------------------------------------------------------
__global__ void prefix_kernel(const int* __restrict__ idxd,
                              int* __restrict__ dest_of, int* __restrict__ meta)
{
  __shared__ int cnt[512];
  int t = threadIdx.x;
  int base = t * 32;
  int local = 0;
  for (int i = 0; i < 32; i++) local += (idxd[base + i] != 0);
  cnt[t] = local;
  __syncthreads();
  for (int off = 1; off < 512; off <<= 1) {
    int v = (t >= off) ? cnt[t - off] : 0;
    __syncthreads();
    cnt[t] += v;
    __syncthreads();
  }
  int run = cnt[t] - local;          // exclusive prefix of True rows
  if (t == 511) meta[0] = cnt[511];  // total True rows
  for (int i = 0; i < 32; i++) {
    if (idxd[base + i] != 0) dest_of[run++] = base + i;
  }
}

// f = sigmoid(gather(fx,d) + gather(fr,r) + gather(fl,l)); fc = f*c0;
// scatter_add fc into c_acc at tree_ids_d.
__global__ __launch_bounds__(256) void fgate_kernel(
    const float* __restrict__ outx, const float* __restrict__ frfl,
    const float* __restrict__ c0,
    const int* __restrict__ idxd, const int* __restrict__ idxr,
    const int* __restrict__ idxl, float* __restrict__ c_acc)
{
  size_t gid = (size_t)blockIdx.x * 256 + threadIdx.x;  // BL * 256 threads
  int m  = (int)(gid >> 8);
  int c4 = (int)(gid & 255) * 4;
  int rowbase = (m >> 9) << 9;
  int jd = rowbase + idxd[m];
  int jr = rowbase + idxr[m];
  int jl = rowbase + idxl[m];
  const float4 fx = *(const float4*)(outx + (size_t)jd * NDIM + 3072 + c4);
  const float4 fr = *(const float4*)(frfl + (size_t)jr * 2048 + c4);
  const float4 fl = *(const float4*)(frfl + (size_t)jl * 2048 + 1024 + c4);
  const float4 cv = *(const float4*)(c0 + (size_t)m * HDIM + c4);
  float4 fc;
  fc.x = sigmoidf_(fx.x + fr.x + fl.x) * cv.x;
  fc.y = sigmoidf_(fx.y + fr.y + fl.y) * cv.y;
  fc.z = sigmoidf_(fx.z + fr.z + fl.z) * cv.z;
  fc.w = sigmoidf_(fx.w + fr.w + fl.w) * cv.w;
  float* dst = c_acc + (size_t)jd * HDIM + c4;
  atomicAdd(dst + 0, fc.x);
  atomicAdd(dst + 1, fc.y);
  atomicAdd(dst + 2, fc.z);
  atomicAdd(dst + 3, fc.w);
}

// Gates + permuted store straight to d_out (h_out | c_out).
__global__ __launch_bounds__(256) void gates_kernel(
    const float* __restrict__ outx, const float* __restrict__ iou_acc,
    const float* __restrict__ c_acc,
    const int* __restrict__ dest_of, const int* __restrict__ meta,
    float* __restrict__ h_out, float* __restrict__ c_out)
{
  size_t gid = (size_t)blockIdx.x * 256 + threadIdx.x;
  int m  = (int)(gid >> 8);
  int c4 = (int)(gid & 255) * 4;
  if (m >= meta[0]) return;             // source rows beyond #True are dropped
  int dest = dest_of[m];
  const float4 ix = *(const float4*)(outx + (size_t)m * NDIM + c4);
  const float4 ox = *(const float4*)(outx + (size_t)m * NDIM + 1024 + c4);
  const float4 ux = *(const float4*)(outx + (size_t)m * NDIM + 2048 + c4);
  const float4 ia = *(const float4*)(iou_acc + (size_t)m * HDIM + c4);
  const float4 ca = *(const float4*)(c_acc + (size_t)m * HDIM + c4);
  float4 cn, hn;
  cn.x = sigmoidf_(ix.x + ia.x) * tanhf(ux.x) + ca.x;
  cn.y = sigmoidf_(ix.y + ia.y) * tanhf(ux.y) + ca.y;
  cn.z = sigmoidf_(ix.z + ia.z) * tanhf(ux.z) + ca.z;
  cn.w = sigmoidf_(ix.w + ia.w) * tanhf(ux.w) + ca.w;
  hn.x = sigmoidf_(ox.x) * tanhf(cn.x);
  hn.y = sigmoidf_(ox.y) * tanhf(cn.y);
  hn.z = sigmoidf_(ox.z) * tanhf(cn.z);
  hn.w = sigmoidf_(ox.w) * tanhf(cn.w);
  *(float4*)(h_out + (size_t)dest * HDIM + c4) = hn;
  *(float4*)(c_out + (size_t)dest * HDIM + c4) = cn;
}

// Rows with tree_ids_d == 0 keep h0/c0.
__global__ __launch_bounds__(256) void copy_untouched_kernel(
    const float* __restrict__ h0, const float* __restrict__ c0,
    const int* __restrict__ idxd,
    float* __restrict__ h_out, float* __restrict__ c_out)
{
  size_t gid = (size_t)blockIdx.x * 256 + threadIdx.x;
  int m  = (int)(gid >> 8);
  int c4 = (int)(gid & 255) * 4;
  if (idxd[m] != 0) return;
  *(float4*)(h_out + (size_t)m * HDIM + c4) = *(const float4*)(h0 + (size_t)m * HDIM + c4);
  *(float4*)(c_out + (size_t)m * HDIM + c4) = *(const float4*)(c0 + (size_t)m * HDIM + c4);
}

// ---------------------------------------------------------------------------
extern "C" void kernel_launch(void* const* d_in, const int* in_sizes, int n_in,
                              void* d_out, int out_size, void* d_ws, size_t ws_size,
                              hipStream_t stream)
{
  const float* x      = (const float*)d_in[0];
  const float* h0     = (const float*)d_in[1];
  const float* c0     = (const float*)d_in[2];
  const float* W_ioux = (const float*)d_in[3];
  const float* W_iouh = (const float*)d_in[4];
  const float* b_iouh = (const float*)d_in[5];
  const float* W_fx   = (const float*)d_in[6];
  const float* W_fh   = (const float*)d_in[7];
  const float* b_fh   = (const float*)d_in[8];
  const int*   idxd   = (const int*)d_in[9];
  const int*   idxr   = (const int*)d_in[10];
  const int*   idxl   = (const int*)d_in[11];

  char* w = (char*)d_ws;
  unsigned short* xb  = (unsigned short*)w;  w += (size_t)BL * KDIM * 2;    // 32 MB
  unsigned short* hb  = (unsigned short*)w;  w += (size_t)BL * KDIM * 2;    // 32 MB
  unsigned short* Wpx = (unsigned short*)w;  w += (size_t)NDIM * KDIM * 2;  // 8 MB
  unsigned short* Wph = (unsigned short*)w;  w += (size_t)NDIM * KDIM * 2;  // 8 MB
  float* biasN        = (float*)w;           w += (size_t)NDIM * 4;
  float* outx         = (float*)w;           w += (size_t)BL * NDIM * 4;    // 256 MB
  float* frfl         = (float*)w;           w += (size_t)BL * 2048 * 4;    // 128 MB
  float* iou_acc      = (float*)w;           w += (size_t)BL * HDIM * 4;    // 64 MB
  float* c_acc        = (float*)w;           w += (size_t)BL * HDIM * 4;    // 64 MB (contiguous after iou_acc)
  int*   dest_of      = (int*)w;             w += (size_t)BL * 4;
  int*   meta         = (int*)w;             w += 256;

  float* h_out = (float*)d_out;
  float* c_out = h_out + (size_t)BL * HDIM;

  dim3 blk(256);

  pack_weights_kernel<<<(NDIM * KDIM) / 256, blk, 0, stream>>>(
      W_ioux, W_iouh, W_fx, W_fh, Wpx, Wph);
  pack_bias_kernel<<<NDIM / 256, blk, 0, stream>>>(b_iouh, b_fh, biasN);
  convert_bf16_kernel<<<(BL * KDIM / 4) / 256, blk, 0, stream>>>(x, h0, xb, hb);

  // zero iou_acc + c_acc (adjacent) every launch
  size_t zf4 = (size_t)BL * HDIM * 2 / 4;
  zero_kernel<<<(unsigned)((zf4 + 255) / 256), blk, 0, stream>>>(iou_acc, zf4);

  prefix_kernel<<<1, 512, 0, stream>>>(idxd, dest_of, meta);

  dim3 ggrid(NDIM / BN, BL / BM);   // 32 x 128 blocks
  gemm_wmma_kernel<<<ggrid, blk, 0, stream>>>(
      xb, Wpx, outx, nullptr, nullptr, nullptr, nullptr, nullptr, 0);
  gemm_wmma_kernel<<<ggrid, blk, 0, stream>>>(
      hb, Wph, nullptr, biasN, idxr, idxl, iou_acc, frfl, 1);

  fgate_kernel<<<(BL * 256) / 256, blk, 0, stream>>>(
      outx, frfl, c0, idxd, idxr, idxl, c_acc);
  gates_kernel<<<(BL * 256) / 256, blk, 0, stream>>>(
      outx, iou_acc, c_acc, dest_of, meta, h_out, c_out);
  copy_untouched_kernel<<<(BL * 256) / 256, blk, 0, stream>>>(
      h0, c0, idxd, h_out, c_out);
}
